// ImprovedGAT_28802050687001
// MI455X (gfx1250) — matrix operand order repaired and checked by reading, not compiled, not run on hardware
//
#include <hip/hip_runtime.h>
#include <hip/hip_bf16.h>

#define N_NODES 50000
#define E_EDGES 800000
#define IN_DIM  256
#define HID     128
#define HEADS   8

typedef __attribute__((ext_vector_type(16))) __bf16 v16bf;
typedef __attribute__((ext_vector_type(8)))  float  v8f;
typedef __bf16 bf16_t;

union Frag32B { v16bf v; uint4 q[2]; };

// ---------------- utility kernels ----------------

__global__ void zero_u32_k(unsigned* __restrict__ p, long long n) {
    long long i = (long long)blockIdx.x * blockDim.x + threadIdx.x;
    long long stride = (long long)gridDim.x * blockDim.x;
    for (; i < n; i += stride) p[i] = 0u;
}

__global__ void cvt_bf16_k(const float* __restrict__ src, bf16_t* __restrict__ dst, long long n) {
    long long i = (long long)blockIdx.x * blockDim.x + threadIdx.x;
    long long stride = (long long)gridDim.x * blockDim.x;
    for (; i < n; i += stride) dst[i] = (bf16_t)src[i];
}

// Pack row-major W[K x Ncols] (f32) into WMMA B-fragment order (bf16):
// fragment (kt, nt): lane l -> col nt*16+(l&15), h=l>>4; element j:
//                    K = kt*32 + 8h + j (j<8), K = kt*32 + 16 + 8h + (j-8) (j>=8)
__global__ void pack_b_k(const float* __restrict__ W, bf16_t* __restrict__ dst, int K, int Ncols) {
    int total = K * Ncols;
    int NT = Ncols >> 4;
    for (int i = blockIdx.x * blockDim.x + threadIdx.x; i < total; i += gridDim.x * blockDim.x) {
        int k = i / Ncols, n = i - k * Ncols;
        int kt = k >> 5, ko = k & 31, nt = n >> 4, no = n & 15;
        int lane = ((ko >> 3) & 1) ? (16 + no) : no;
        int j    = (ko & 7) + ((ko >= 16) ? 8 : 0);
        dst[(((size_t)(kt * NT + nt)) * 32 + lane) * 16 + j] = (bf16_t)W[i];
    }
}

// ---------------- WMMA GEMM: C[M x 128] = A[M x KDIM] * Bpk ----------------
// MODE 0: plain   MODE 1: +bias, leaky   MODE 2: + add[] + bias[] + bias2[]
template<int KDIM, int MODE>
__global__ __launch_bounds__(256)
void gemm_wmma_k(const bf16_t* __restrict__ A, const bf16_t* __restrict__ Bpk,
                 float* __restrict__ C,
                 const float* __restrict__ bias,
                 const float* __restrict__ add,
                 const float* __restrict__ bias2) {
    const int lane = threadIdx.x & 31;
    const int nt   = threadIdx.x >> 5;      // 8 waves -> 8 N tiles
    const int mt   = blockIdx.x;
    const int NT   = HID / 16;
    const int row0 = mt * 16;
    const int half = lane >> 4;
    const int l16  = lane & 15;

    const bf16_t* Arow = A + (size_t)(row0 + l16) * KDIM;
    v8f acc = {0.f, 0.f, 0.f, 0.f, 0.f, 0.f, 0.f, 0.f};

#pragma unroll
    for (int kt = 0; kt < KDIM / 32; ++kt) {
        Frag32B a, b;
        const int kbase = kt * 32 + (half ? 8 : 0);
        a.q[0] = *(const uint4*)(Arow + kbase);
        a.q[1] = *(const uint4*)(Arow + kbase + 16);
        const bf16_t* bp = Bpk + ((size_t)(kt * NT + nt) * 32 + lane) * 16;
        b.q[0] = *(const uint4*)(bp);
        b.q[1] = *(const uint4*)(bp + 8);
        acc = __builtin_amdgcn_wmma_f32_16x16x32_bf16(false, a.v, false, b.v,
                                                      (short)0, acc, false, false);
    }

    const int col = nt * 16 + l16;
    const size_t rbase = (size_t)(row0 + half * 8);
    float* Cp = C + rbase * HID + col;
#pragma unroll
    for (int r = 0; r < 8; ++r) {
        float v = acc[r];
        if (MODE == 1) { v += bias[col]; v = (v >= 0.f) ? v : 0.2f * v; }
        if (MODE == 2) { v += add[(rbase + r) * HID + col] + bias[col] + bias2[col]; }
        Cp[(size_t)r * HID] = v;
    }
}

// ---------------- batchnorm helpers ----------------

__global__ __launch_bounds__(128)
void col_stats_k(const float* __restrict__ H, float* __restrict__ gsum,
                 float* __restrict__ gsq, int nrows) {
    int c = threadIdx.x;
    float s = 0.f, q = 0.f;
    for (int r = blockIdx.x; r < nrows; r += gridDim.x) {
        float v = H[(size_t)r * HID + c];
        s += v; q += v * v;
    }
    atomicAdd(&gsum[c], s);
    atomicAdd(&gsq[c],  q);
}

__global__ void bn_finalize_k(const float* __restrict__ gsum, const float* __restrict__ gsq,
                              float* __restrict__ mean, float* __restrict__ rstd, float inv_n) {
    int c = threadIdx.x;  // 128 threads
    float m = gsum[c] * inv_n;
    float v = gsq[c] * inv_n - m * m;
    mean[c] = m;
    rstd[c] = rsqrtf(fmaxf(v, 0.f) + 1e-5f);
}

__global__ void bn_apply_cvt_k(const float* __restrict__ H, const float* __restrict__ mean,
                               const float* __restrict__ rstd, const float* __restrict__ gamma,
                               const float* __restrict__ beta, bf16_t* __restrict__ out, long long n) {
    long long i = (long long)blockIdx.x * blockDim.x + threadIdx.x;
    long long stride = (long long)gridDim.x * blockDim.x;
    for (; i < n; i += stride) {
        int c = (int)(i & (HID - 1));
        out[i] = (bf16_t)((H[i] - mean[c]) * rstd[c] * gamma[c] + beta[c]);
    }
}

// ---------------- attention coefficient kernels ----------------

__global__ void att_heads_k(const float* __restrict__ T, const float* __restrict__ att_s,
                            const float* __restrict__ att_d, float* __restrict__ as_,
                            float* __restrict__ ad_) {
    int i = blockIdx.x * blockDim.x + threadIdx.x;   // n*HEADS + h
    if (i >= N_NODES * HEADS) return;
    int h = i & (HEADS - 1);
    const float* t = T + (size_t)(i >> 3) * HID + h * 16;
    float s = 0.f, d = 0.f;
#pragma unroll
    for (int c = 0; c < 16; ++c) {
        float v = t[c];
        s += v * att_s[h * 16 + c];
        d += v * att_d[h * 16 + c];
    }
    as_[i] = s; ad_[i] = d;
}

__global__ __launch_bounds__(128)
void att_single_k(const float* __restrict__ T, const float* __restrict__ att_s,
                  const float* __restrict__ att_d, float* __restrict__ as_,
                  float* __restrict__ ad_) {
    __shared__ float ls[128], ld_[128];
    int n = blockIdx.x, c = threadIdx.x;
    float v = T[(size_t)n * HID + c];
    ls[c]  = v * att_s[c];
    ld_[c] = v * att_d[c];
    __syncthreads();
    for (int s = 64; s > 0; s >>= 1) {
        if (c < s) { ls[c] += ls[c + s]; ld_[c] += ld_[c + s]; }
        __syncthreads();
    }
    if (c == 0) { as_[n] = ls[0]; ad_[n] = ld_[0]; }
}

// ---------------- edge-phase kernels ----------------

__device__ inline unsigned fenc(float f) {
    unsigned u = __float_as_uint(f);
    return (u & 0x80000000u) ? ~u : (u | 0x80000000u);
}
__device__ inline float fdec(unsigned u) {
    return (u & 0x80000000u) ? __uint_as_float(u & 0x7fffffffu) : __uint_as_float(~u);
}

__global__ void edge_e_k(const long long* __restrict__ src, const long long* __restrict__ dst,
                         const float* __restrict__ as_, const float* __restrict__ ad_,
                         float* __restrict__ e, unsigned* __restrict__ key,
                         int heads, long long total) {
    long long i = (long long)blockIdx.x * blockDim.x + threadIdx.x;
    long long stride = (long long)gridDim.x * blockDim.x;
    for (; i < total; i += stride) {
        int h = (int)(i % heads);
        long long ed = i / heads;
        int s = (int)src[ed], d = (int)dst[ed];
        float v = as_[(size_t)s * heads + h] + ad_[(size_t)d * heads + h];
        v = (v >= 0.f) ? v : 0.2f * v;
        e[i] = v;
        atomicMax(&key[(size_t)d * heads + h], fenc(v));
    }
}

__global__ void m_fin_k(const unsigned* __restrict__ key, float* __restrict__ m, int n) {
    int i = blockIdx.x * blockDim.x + threadIdx.x;
    if (i >= n) return;
    unsigned k = key[i];
    m[i] = k ? fdec(k) : 0.f;   // nodes with no in-edges: max -> 0 (matches reference)
}

__global__ void ex_den_k(const long long* __restrict__ dst, const float* __restrict__ m,
                         float* __restrict__ e, float* __restrict__ den,
                         int heads, long long total) {
    long long i = (long long)blockIdx.x * blockDim.x + threadIdx.x;
    long long stride = (long long)gridDim.x * blockDim.x;
    for (; i < total; i += stride) {
        int h = (int)(i % heads);
        long long ed = i / heads;
        int d = (int)dst[ed];
        float ex = expf(e[i] - m[(size_t)d * heads + h]);
        e[i] = ex;
        atomicAdd(&den[(size_t)d * heads + h], ex);
    }
}

__global__ void aggregate_k(const long long* __restrict__ src, const long long* __restrict__ dst,
                            const float* __restrict__ ex, const float* __restrict__ den,
                            const float* __restrict__ T, float* __restrict__ out, int heads) {
    long long total = (long long)E_EDGES * HID;
    long long i = (long long)blockIdx.x * blockDim.x + threadIdx.x;
    long long stride = (long long)gridDim.x * blockDim.x;
    for (; i < total; i += stride) {
        int c = (int)(i & (HID - 1));
        long long ed = i >> 7;
        int h = (heads == 1) ? 0 : (c >> 4);
        int s = (int)src[ed], d = (int)dst[ed];
        float alpha = ex[ed * heads + h] / fmaxf(den[(size_t)d * heads + h], 1e-16f);
        atomicAdd(&out[(size_t)d * HID + c], alpha * T[(size_t)s * HID + c]);
    }
}

__global__ void post_leaky_k(const float* __restrict__ out1, const float* __restrict__ bias,
                             float* __restrict__ skip, bf16_t* __restrict__ skipb, long long n) {
    long long i = (long long)blockIdx.x * blockDim.x + threadIdx.x;
    long long stride = (long long)gridDim.x * blockDim.x;
    for (; i < n; i += stride) {
        int c = (int)(i & (HID - 1));
        float v = out1[i] + bias[c];
        v = (v >= 0.f) ? v : 0.2f * v;
        skip[i]  = v;
        skipb[i] = (bf16_t)v;
    }
}

__global__ __launch_bounds__(128)
void bn_norm_out_k(const float* __restrict__ H, const float* __restrict__ mean,
                   const float* __restrict__ rstd, const float* __restrict__ gamma,
                   const float* __restrict__ beta, float* __restrict__ out) {
    __shared__ float sh[128];
    int n = blockIdx.x, c = threadIdx.x;
    float v = (H[(size_t)n * HID + c] - mean[c]) * rstd[c] * gamma[c] + beta[c];
    sh[c] = v * v;
    __syncthreads();
    for (int s = 64; s > 0; s >>= 1) {
        if (c < s) sh[c] += sh[c + s];
        __syncthreads();
    }
    float norm = fmaxf(sqrtf(sh[0]), 1e-12f);
    out[(size_t)n * HID + c] = v / norm;
}

// ---------------- host launcher ----------------

extern "C" void kernel_launch(void* const* d_in, const int* in_sizes, int n_in,
                              void* d_out, int out_size, void* d_ws, size_t ws_size,
                              hipStream_t stream) {
    (void)in_sizes; (void)n_in; (void)out_size; (void)ws_size;

    const float* x      = (const float*)d_in[0];
    const float* W_in   = (const float*)d_in[1];
    const float* b_in   = (const float*)d_in[2];
    const float* gamma1 = (const float*)d_in[3];
    const float* beta1  = (const float*)d_in[4];
    const float* W1     = (const float*)d_in[5];
    const float* att_s1 = (const float*)d_in[6];
    const float* att_d1 = (const float*)d_in[7];
    const float* bias1  = (const float*)d_in[8];
    const float* W2     = (const float*)d_in[9];
    const float* att_s2 = (const float*)d_in[10];
    const float* att_d2 = (const float*)d_in[11];
    const float* bias2  = (const float*)d_in[12];
    const float* W_skip = (const float*)d_in[13];
    const float* b_skip = (const float*)d_in[14];
    const float* gamma2 = (const float*)d_in[15];
    const float* beta2  = (const float*)d_in[16];
    const long long* ei = (const long long*)d_in[17];
    const long long* srcv = ei;
    const long long* dstv = ei + E_EDGES;

    char* ws = (char*)d_ws;
    size_t off = 0;
    auto take = [&](size_t bytes) -> char* {
        char* p = ws + off;
        off += (bytes + 255) & ~(size_t)255;
        return p;
    };

    bf16_t* wpk_in = (bf16_t*)take((size_t)IN_DIM * HID * 2);
    bf16_t* wpk_1  = (bf16_t*)take((size_t)HID * HID * 2);
    bf16_t* wpk_2  = (bf16_t*)take((size_t)HID * HID * 2);
    bf16_t* wpk_sk = (bf16_t*)take((size_t)HID * HID * 2);

    char*  bufA  = take((size_t)N_NODES * IN_DIM * 2);  // xb (bf16), later t1 (f32, same size)
    bf16_t* xb   = (bf16_t*)bufA;
    float*  t1   = (float*)bufA;

    float* h1    = (float*)take((size_t)N_NODES * HID * 4);  // later t2
    float* t2    = h1;

    bf16_t* hbn  = (bf16_t*)take((size_t)N_NODES * HID * 2);

    float*    as1  = (float*)take((size_t)N_NODES * HEADS * 4);
    float*    ad1  = (float*)take((size_t)N_NODES * HEADS * 4);
    unsigned* key1 = (unsigned*)take((size_t)N_NODES * HEADS * 4);
    float*    m1   = (float*)take((size_t)N_NODES * HEADS * 4);
    float*    den1 = (float*)take((size_t)N_NODES * HEADS * 4);

    float* e1   = (float*)take((size_t)E_EDGES * HEADS * 4);  // later e2 (first E floats)
    float* e2   = e1;
    float* out1 = (float*)take((size_t)N_NODES * HID * 4);    // later h2
    float* h2   = out1;
    float* skip = (float*)take((size_t)N_NODES * HID * 4);
    bf16_t* skipb = (bf16_t*)take((size_t)N_NODES * HID * 2);
    float* out2 = (float*)take((size_t)N_NODES * HID * 4);

    float*    as2  = (float*)take((size_t)N_NODES * 4);
    float*    ad2  = (float*)take((size_t)N_NODES * 4);
    unsigned* key2 = (unsigned*)take((size_t)N_NODES * 4);
    float*    m2   = (float*)take((size_t)N_NODES * 4);
    float*    den2 = (float*)take((size_t)N_NODES * 4);

    float* stats = (float*)take(1024 * 4);
    float *sum1 = stats, *sq1 = stats + 128, *mean1 = stats + 256, *rstd1 = stats + 384;
    float *sum2 = stats + 512, *sq2 = stats + 640, *mean2 = stats + 768, *rstd2 = stats + 896;

    const int TB = 256, GB = 4096;
    const int MT = N_NODES / 16;  // 3125 row tiles

    // re-zero all accumulators (graph replays do not re-poison)
    zero_u32_k<<<256, TB, 0, stream>>>(key1, (long long)N_NODES * HEADS);
    zero_u32_k<<<256, TB, 0, stream>>>((unsigned*)den1, (long long)N_NODES * HEADS);
    zero_u32_k<<<GB, TB, 0, stream>>>((unsigned*)out1, (long long)N_NODES * HID);
    zero_u32_k<<<64, TB, 0, stream>>>(key2, N_NODES);
    zero_u32_k<<<64, TB, 0, stream>>>((unsigned*)den2, N_NODES);
    zero_u32_k<<<GB, TB, 0, stream>>>((unsigned*)out2, (long long)N_NODES * HID);
    zero_u32_k<<<4, TB, 0, stream>>>((unsigned*)stats, 1024);

    // weight packing + activation conversion
    pack_b_k<<<64, TB, 0, stream>>>(W_in, wpk_in, IN_DIM, HID);
    pack_b_k<<<32, TB, 0, stream>>>(W1, wpk_1, HID, HID);
    pack_b_k<<<32, TB, 0, stream>>>(W2, wpk_2, HID, HID);
    pack_b_k<<<32, TB, 0, stream>>>(W_skip, wpk_sk, HID, HID);
    cvt_bf16_k<<<GB, TB, 0, stream>>>(x, xb, (long long)N_NODES * IN_DIM);

    // h1 = leaky(x @ W_in + b_in)
    gemm_wmma_k<IN_DIM, 1><<<MT, TB, 0, stream>>>(xb, wpk_in, h1, b_in, nullptr, nullptr);

    // BN1 -> hbn (bf16)
    col_stats_k<<<512, 128, 0, stream>>>(h1, sum1, sq1, N_NODES);
    bn_finalize_k<<<1, 128, 0, stream>>>(sum1, sq1, mean1, rstd1, 1.0f / N_NODES);
    bn_apply_cvt_k<<<GB, TB, 0, stream>>>(h1, mean1, rstd1, gamma1, beta1, hbn,
                                          (long long)N_NODES * HID);

    // conv1: t1 = hbn @ W1 ; attention ; softmax-aggregate
    gemm_wmma_k<HID, 0><<<MT, TB, 0, stream>>>(hbn, wpk_1, t1, nullptr, nullptr, nullptr);
    att_heads_k<<<(N_NODES * HEADS + TB - 1) / TB, TB, 0, stream>>>(t1, att_s1, att_d1, as1, ad1);
    edge_e_k<<<GB, TB, 0, stream>>>(srcv, dstv, as1, ad1, e1, key1, HEADS,
                                    (long long)E_EDGES * HEADS);
    m_fin_k<<<(N_NODES * HEADS + TB - 1) / TB, TB, 0, stream>>>(key1, m1, N_NODES * HEADS);
    ex_den_k<<<GB, TB, 0, stream>>>(dstv, m1, e1, den1, HEADS, (long long)E_EDGES * HEADS);
    aggregate_k<<<2 * GB, TB, 0, stream>>>(srcv, dstv, e1, den1, t1, out1, HEADS);

    // skip = leaky(out1 + bias1)
    post_leaky_k<<<GB, TB, 0, stream>>>(out1, bias1, skip, skipb, (long long)N_NODES * HID);

    // conv2 (heads=1): t2 = skip @ W2 ; attention ; aggregate
    gemm_wmma_k<HID, 0><<<MT, TB, 0, stream>>>(skipb, wpk_2, t2, nullptr, nullptr, nullptr);
    att_single_k<<<N_NODES, 128, 0, stream>>>(t2, att_s2, att_d2, as2, ad2);
    edge_e_k<<<GB, TB, 0, stream>>>(srcv, dstv, as2, ad2, e2, key2, 1, (long long)E_EDGES);
    m_fin_k<<<(N_NODES + TB - 1) / TB, TB, 0, stream>>>(key2, m2, N_NODES);
    ex_den_k<<<GB, TB, 0, stream>>>(dstv, m2, e2, den2, 1, (long long)E_EDGES);
    aggregate_k<<<2 * GB, TB, 0, stream>>>(srcv, dstv, e2, den2, t2, out2, 1);

    // h2 = (skip @ W_skip) + out2 + bias2 + b_skip  (fused epilogue)
    gemm_wmma_k<HID, 2><<<MT, TB, 0, stream>>>(skipb, wpk_sk, h2, bias2, out2, b_skip);

    // BN2 + row L2 normalize -> output
    col_stats_k<<<512, 128, 0, stream>>>(h2, sum2, sq2, N_NODES);
    bn_finalize_k<<<1, 128, 0, stream>>>(sum2, sq2, mean2, rstd2, 1.0f / N_NODES);
    bn_norm_out_k<<<N_NODES, 128, 0, stream>>>(h2, mean2, rstd2, gamma2, beta2, (float*)d_out);
}
